// GraphAttentionV2Network_51797305590029
// MI455X (gfx1250) — matrix operand order repaired and checked by reading, not compiled, not run on hardware
//
#include <hip/hip_runtime.h>

// ---------------- problem constants ----------------
#define NN  50000
#define EE  400000
#define GG  500
#define HH  4
#define CC  64
#define HC  256   // H*C
#define IND 9
#define EDD 3
#define NEG 0.2f
#define BNEPS 1e-5f
#define MT  (NN / 16)          // 3125 M-tiles (NN is a multiple of 16)
#define BPK_ELEMS (2 * 16 * 32 * 16)   // packed B: [kt][ntile][lane][elem] bf16

// ---------------- WMMA vector types ----------------
typedef __attribute__((ext_vector_type(16))) __bf16 v16bf;
typedef __attribute__((ext_vector_type(8)))  __bf16 v8bf;
typedef __attribute__((ext_vector_type(8)))  float  v8f;

__device__ __forceinline__ unsigned short f2bf(float f) {
  unsigned int u = __float_as_uint(f);
  unsigned int r = (u + 0x7FFFu + ((u >> 16) & 1u)) >> 16;
  return (unsigned short)r;
}

// ordered-uint encoding of float for atomicMax-based segment max
__device__ __forceinline__ unsigned int f2ord(float f) {
  unsigned int u = __float_as_uint(f);
  return (u & 0x80000000u) ? ~u : (u | 0x80000000u);
}
__device__ __forceinline__ float ord2f(unsigned int u) {
  return __uint_as_float((u & 0x80000000u) ? (u ^ 0x80000000u) : ~u);
}

// ---------------- generic zero fill ----------------
__global__ void k_zero(float* __restrict__ p, long n) {
  long t = (long)blockIdx.x * blockDim.x + threadIdx.x;
  if (t < n) p[t] = 0.0f;
}

// ---------------- layer-0 node transforms (K=9, scalar) ----------------
__global__ void k_layer0_xlr(const float* __restrict__ x,
                             const float* __restrict__ W0l, const float* __restrict__ b0l,
                             const float* __restrict__ W0r,
                             float* __restrict__ xl, float* __restrict__ xr) {
  long t = (long)blockIdx.x * blockDim.x + threadIdx.x;
  if (t >= (long)NN * HC) return;
  int n = (int)(t >> 8);
  int oc = (int)(t & 255);
  float al = b0l[oc], ar = 0.0f;
#pragma unroll
  for (int k = 0; k < IND; ++k) {
    float xv = x[n * IND + k];
    al += xv * W0l[k * HC + oc];
    ar += xv * W0r[k * HC + oc];
  }
  xl[t] = al;
  xr[t] = ar;
}

// ---------------- pack 64x256 f32 weight into WMMA-B bf16 lane layout ----
// Bpk layout: [kt(2)][ntile(16)][lane(32)][elem(16)] bf16
// lane l (0..15): col = ntile*16 + l, K = kt*32 + j      (j = 0..15)
// lane l (16..31): col = ntile*16 + (l&15), K = kt*32 + 16 + j
__global__ void k_pack_w(const float* __restrict__ W, unsigned short* __restrict__ Bpk) {
  int t = blockIdx.x * blockDim.x + threadIdx.x;
  if (t >= BPK_ELEMS) return;
  int j  = t & 15;
  int l  = (t >> 4) & 31;
  int nt = (t >> 9) & 15;
  int kt = t >> 13;
  int k  = kt * 32 + ((l & 16) ? 16 : 0) + j;
  int col = nt * 16 + (l & 15);
  Bpk[t] = f2bf(W[k * HC + col]);
}

// ---------------- WMMA GEMM: C[N x 256] = A[N x 64](bf16) @ Bpk + bias ---
// One wave owns a full 16-row block and sweeps all 16 column tiles:
// A fragments loaded once into VGPRs, B fed from LDS, 32 WMMAs per wave.
// `bias` is always a valid 256-float vector (zeros for the no-bias GEMM),
// so there is no pointer-null branch in the inner loop.
__global__ void __launch_bounds__(256)
k_wmma_gemm(const unsigned short* __restrict__ Abf,   // NN x 64 bf16, row major
            const unsigned short* __restrict__ Bpk,   // packed per k_pack_w
            const float* __restrict__ bias,           // 256 floats, always valid
            float* __restrict__ Cout) {
  __shared__ __align__(32) unsigned short Bsh[BPK_ELEMS];   // 32 KB of 320 KB WGP LDS

  int wave = (blockIdx.x * blockDim.x + threadIdx.x) >> 5;
  int l = threadIdx.x & 31;
  int row = wave * 16 + (l & 15);
  int kh = l >> 4;                 // 0 for lanes 0-15, 1 for lanes 16-31

  // prefetch this wave's A row so the fetch overlaps the LDS staging
  if (wave < MT) {
    __builtin_prefetch(Abf + (size_t)row * CC, 0, 0);
  }

  // cooperative stage of packed B into LDS: 256 threads x 8 uint4
  {
    const uint4* gs = (const uint4*)Bpk;
    uint4* ls = (uint4*)Bsh;
    int t = threadIdx.x;
#pragma unroll
    for (int i = 0; i < 8; ++i) ls[t + 256 * i] = gs[t + 256 * i];
  }
  __syncthreads();
  if (wave >= MT) return;

  // load both A fragments (K = 0..31, 32..63) once
  v16bf av[2];
#pragma unroll
  for (int kt = 0; kt < 2; ++kt) {
    // A k order used here (contiguous, same as Bpk): lanes 0-15 hold K = kt*32 + {0..15},
    //                          lanes 16-31 hold K = kt*32 + {16..31}
    int base_k = kt * 32 + kh * 16;
    uint4 a0 = *(const uint4*)(Abf + (size_t)row * CC + base_k);
    uint4 a1 = *(const uint4*)(Abf + (size_t)row * CC + base_k + 8);
    v8bf lo = __builtin_bit_cast(v8bf, a0);
    v8bf hi = __builtin_bit_cast(v8bf, a1);
#pragma unroll
    for (int j = 0; j < 8; ++j) { av[kt][j] = lo[j]; av[kt][j + 8] = hi[j]; }
  }

  int colbase = l & 15;
  int rbase = wave * 16 + kh * 8;   // D layout: VGPR v -> M = rbase + v
#pragma unroll
  for (int nt = 0; nt < 16; ++nt) {
    v8f acc = {};
#pragma unroll
    for (int kt = 0; kt < 2; ++kt) {
      v16bf bv = *(const v16bf*)(Bsh + (((kt * 16 + nt) * 32 + l) << 4));
      acc = __builtin_amdgcn_wmma_f32_16x16x32_bf16(false, av[kt], false, bv,
                                                    (short)0, acc, false, false);
    }
    int col = nt * 16 + colbase;
    float badd = bias[col];
#pragma unroll
    for (int v = 0; v < 8; ++v) {
      Cout[(size_t)(rbase + v) * HC + col] = acc[v] + badd;
    }
  }
}

// ---------------- edge attention logits: one wave per edge ---------------
// s[e,h] = sum_c leakyrelu(xl[src,h,c]+xr[dst,h,c]+ee[e,h,c]) * att[h,c]
// ee recomputed inline from ea @ We (K=3). atomicMax segment max on dst.
__global__ void __launch_bounds__(256)
k_edge_att(const int* __restrict__ src, const int* __restrict__ dst,
           const float* __restrict__ ea, const float* __restrict__ We,
           const float* __restrict__ att,
           const float* __restrict__ xl, const float* __restrict__ xr,
           float* __restrict__ s, unsigned int* __restrict__ smax) {
  int gid = blockIdx.x * blockDim.x + threadIdx.x;
  int e = gid >> 5;
  int l = gid & 31;
  if (e >= EE) return;
  int sn = src[e], dn = dst[e];
  float e0 = ea[e * EDD + 0], e1 = ea[e * EDD + 1], e2 = ea[e * EDD + 2];
  int cb = l * 8;
  float4 A0 = *(const float4*)(xl + (long)sn * HC + cb);
  float4 A1 = *(const float4*)(xl + (long)sn * HC + cb + 4);
  float4 R0 = *(const float4*)(xr + (long)dn * HC + cb);
  float4 R1 = *(const float4*)(xr + (long)dn * HC + cb + 4);
  float la[8] = {A0.x, A0.y, A0.z, A0.w, A1.x, A1.y, A1.z, A1.w};
  float lr[8] = {R0.x, R0.y, R0.z, R0.w, R1.x, R1.y, R1.z, R1.w};
  float acc = 0.0f;
#pragma unroll
  for (int j = 0; j < 8; ++j) {
    int c = cb + j;
    float eec = e0 * We[c] + e1 * We[HC + c] + e2 * We[2 * HC + c];
    float m = la[j] + lr[j] + eec;
    m = (m > 0.0f) ? m : NEG * m;
    acc += m * att[c];
  }
  // reduce over the 8 lanes of this head (lanes [h*8, h*8+8))
  acc += __shfl_xor(acc, 1, 32);
  acc += __shfl_xor(acc, 2, 32);
  acc += __shfl_xor(acc, 4, 32);
  if ((l & 7) == 0) {
    int h = l >> 3;
    s[(long)e * HH + h] = acc;
    atomicMax(&smax[(long)dn * HH + h], f2ord(acc));
  }
}

// ---------------- softmax numerator + denominator ------------------------
__global__ void k_edge_exp(const int* __restrict__ dst,
                           const float* __restrict__ s,
                           const unsigned int* __restrict__ smax,
                           float* __restrict__ ex, float* __restrict__ den) {
  long t = (long)blockIdx.x * blockDim.x + threadIdx.x;
  if (t >= (long)EE * HH) return;
  int e = (int)(t >> 2);
  int h = (int)(t & 3);
  int dn = dst[e];
  float sm = ord2f(smax[(long)dn * HH + h]);
  float v = __expf(s[t] - sm);
  ex[t] = v;
  atomicAdd(&den[(long)dn * HH + h], v);
}

// ---------------- weighted scatter: acc[dst] += alpha * xl[src] ----------
__global__ void __launch_bounds__(256)
k_edge_scatter(const int* __restrict__ src, const int* __restrict__ dst,
               const float* __restrict__ xl, const float* __restrict__ ex,
               const float* __restrict__ den, float* __restrict__ acc) {
  int gid = blockIdx.x * blockDim.x + threadIdx.x;
  int e = gid >> 5;
  int l = gid & 31;
  if (e >= EE) return;
  int sn = src[e], dn = dst[e];
  int h = l >> 3;
  int cb = l * 8;
  float alpha = ex[(long)e * HH + h] / den[(long)dn * HH + h];
  const float* p = xl + (long)sn * HC + cb;
  float* q = acc + (long)dn * HC + cb;
  float4 P0 = *(const float4*)(p);
  float4 P1 = *(const float4*)(p + 4);
  float pv[8] = {P0.x, P0.y, P0.z, P0.w, P1.x, P1.y, P1.z, P1.w};
#pragma unroll
  for (int j = 0; j < 8; ++j) atomicAdd(&q[j], pv[j] * alpha);
}

// ---------------- mean-over-heads + bias + BN + ELU (+residual) ----------
__global__ void k_node_final(const float* __restrict__ acc, const float* __restrict__ cbias,
                             const float* __restrict__ bng, const float* __restrict__ bnb,
                             const float* __restrict__ bnm, const float* __restrict__ bnv,
                             const float* __restrict__ hprev,   // nullptr for layer 0
                             float* __restrict__ hout, unsigned short* __restrict__ hbf) {
  long t = (long)blockIdx.x * blockDim.x + threadIdx.x;
  if (t >= (long)NN * CC) return;
  int n = (int)(t >> 6);
  int c = (int)(t & 63);
  const float* a = acc + (long)n * HC + c;
  float v = 0.25f * (a[0] + a[CC] + a[2 * CC] + a[3 * CC]) + cbias[c];
  v = (v - bnm[c]) * (bng[c] * rsqrtf(bnv[c] + BNEPS)) + bnb[c];
  v = (v > 0.0f) ? v : (__expf(v) - 1.0f);
  if (hprev) v += hprev[t];
  hout[t] = v;
  hbf[t] = f2bf(v);
}

// ---------------- global mean pool (sum + count) -------------------------
__global__ void k_pool(const float* __restrict__ h, const int* __restrict__ batch,
                       float* __restrict__ pooled, float* __restrict__ cnt) {
  long t = (long)blockIdx.x * blockDim.x + threadIdx.x;
  if (t >= (long)NN * CC) return;
  int n = (int)(t >> 6);
  int c = (int)(t & 63);
  int g = batch[n];
  atomicAdd(&pooled[(long)g * CC + c], h[t]);
  if (c == 0) atomicAdd(&cnt[g], 1.0f);
}

// ---------------- MLP head: one wave per graph ---------------------------
__global__ void k_head(const float* __restrict__ pooled, const float* __restrict__ cnt,
                       const float* __restrict__ Wo1, const float* __restrict__ bo1,
                       const float* __restrict__ Wo2, const float* __restrict__ bo2,
                       float* __restrict__ out) {
  int gid = blockIdx.x * blockDim.x + threadIdx.x;
  int g = gid >> 5;
  int l = gid & 31;
  if (g >= GG) return;
  float inv = 1.0f / fmaxf(cnt[g], 1.0f);
  float hid = bo1[l];
  for (int c = 0; c < CC; ++c)
    hid += pooled[(long)g * CC + c] * inv * Wo1[c * 32 + l];
  hid = (hid > 0.0f) ? hid : (__expf(hid) - 1.0f);
  float o = hid * Wo2[l];
  o += __shfl_xor(o, 1, 32);
  o += __shfl_xor(o, 2, 32);
  o += __shfl_xor(o, 4, 32);
  o += __shfl_xor(o, 8, 32);
  o += __shfl_xor(o, 16, 32);
  if (l == 0) out[g] = o + bo2[0];
}

// ========================= host launcher =================================
extern "C" void kernel_launch(void* const* d_in, const int* in_sizes, int n_in,
                              void* d_out, int out_size, void* d_ws, size_t ws_size,
                              hipStream_t stream) {
  const float* x    = (const float*)d_in[0];
  const int*   ei   = (const int*)  d_in[1];
  const int*   batch= (const int*)  d_in[2];
  const float* ea   = (const float*)d_in[3];
  const float* W0l  = (const float*)d_in[4];
  const float* b0l  = (const float*)d_in[5];
  const float* W0r  = (const float*)d_in[6];
  const float* W0e  = (const float*)d_in[7];
  const float* a0   = (const float*)d_in[8];
  const float* c0   = (const float*)d_in[9];
  const float* Wl   = (const float*)d_in[10];
  const float* bl   = (const float*)d_in[11];
  const float* Wr   = (const float*)d_in[12];
  const float* Wecv = (const float*)d_in[13];
  const float* aatt = (const float*)d_in[14];
  const float* cb   = (const float*)d_in[15];
  const float* bng  = (const float*)d_in[16];
  const float* bnb  = (const float*)d_in[17];
  const float* bnm  = (const float*)d_in[18];
  const float* bnv  = (const float*)d_in[19];
  const float* Wo1  = (const float*)d_in[20];
  const float* bo1  = (const float*)d_in[21];
  const float* Wo2  = (const float*)d_in[22];
  const float* bo2  = (const float*)d_in[23];
  const int* src  = ei;
  const int* dstp = ei + EE;

  // ---- workspace carve-out (256B aligned slabs) ----
  char* ws = (char*)d_ws;
  size_t off = 0;
  auto carve = [&](size_t bytes) -> void* {
    off = (off + 255) & ~(size_t)255;
    void* p = ws + off;
    off += bytes;
    return p;
  };
  float*          xl     = (float*)carve((size_t)NN * HC * 4);
  float*          xr     = (float*)carve((size_t)NN * HC * 4);
  float*          accb   = (float*)carve((size_t)NN * HC * 4);
  float*          hbuf   = (float*)carve((size_t)NN * CC * 4);
  unsigned short* hbf    = (unsigned short*)carve((size_t)NN * CC * 2);
  float*          sbuf   = (float*)carve((size_t)EE * HH * 4);
  float*          exbuf  = (float*)carve((size_t)EE * HH * 4);
  unsigned int*   smax   = (unsigned int*)carve((size_t)NN * HH * 4);
  float*          den    = (float*)carve((size_t)NN * HH * 4);
  unsigned short* BpkL   = (unsigned short*)carve(BPK_ELEMS * 2);
  unsigned short* BpkR   = (unsigned short*)carve(BPK_ELEMS * 2);
  float*          zbias  = (float*)carve((size_t)HC * 4);   // all-zero bias vector
  float*          pooled = (float*)carve((size_t)GG * CC * 4);
  float*          cnt    = (float*)carve((size_t)GG * 4);

  const int T = 256;
  const long nNodeFeat = (long)NN * HC;     // 12.8M
  const long nNodeC    = (long)NN * CC;     // 3.2M
  const long nEdgeH    = (long)EE * HH;     // 1.6M
  const int  gNodeFeat = (int)((nNodeFeat + T - 1) / T);
  const int  gNodeC    = (int)((nNodeC + T - 1) / T);
  const int  gEdgeWave = (int)(((long)EE * 32 + T - 1) / T);   // wave per edge
  const int  gEdgeH    = (int)((nEdgeH + T - 1) / T);
  const int  gGemm     = (int)(((long)MT * 32 + T - 1) / T);   // wave per 16-row block

  // zero bias vector for the no-bias GEMM (xr = h @ Wr)
  k_zero<<<1, T, 0, stream>>>(zbias, HC);

  auto run_edge_phase = [&](const float* We_l, const float* att_l) {
    k_zero<<<gNodeFeat, T, 0, stream>>>(accb, nNodeFeat);
    k_zero<<<(int)(((long)NN * HH + T - 1) / T), T, 0, stream>>>(den, (long)NN * HH);
    k_zero<<<(int)(((long)NN * HH + T - 1) / T), T, 0, stream>>>((float*)smax, (long)NN * HH);
    k_edge_att<<<gEdgeWave, T, 0, stream>>>(src, dstp, ea, We_l, att_l, xl, xr, sbuf, smax);
    k_edge_exp<<<gEdgeH, T, 0, stream>>>(dstp, sbuf, smax, exbuf, den);
    k_edge_scatter<<<gEdgeWave, T, 0, stream>>>(src, dstp, xl, exbuf, den, accb);
  };

  // ---- layer 0 (input GATv2, K=9 scalar transforms) ----
  k_layer0_xlr<<<gNodeFeat, T, 0, stream>>>(x, W0l, b0l, W0r, xl, xr);
  run_edge_phase(W0e, a0);
  k_node_final<<<gNodeC, T, 0, stream>>>(accb, c0, bng, bnb, bnm, bnv,
                                         nullptr, hbuf, hbf);

  // ---- layers 1..3 (residual GATv2, WMMA GEMMs) ----
  for (int i = 0; i < 3; ++i) {
    k_pack_w<<<(BPK_ELEMS + T - 1) / T, T, 0, stream>>>(Wl + (size_t)i * CC * HC, BpkL);
    k_pack_w<<<(BPK_ELEMS + T - 1) / T, T, 0, stream>>>(Wr + (size_t)i * CC * HC, BpkR);
    k_wmma_gemm<<<gGemm, T, 0, stream>>>(hbf, BpkL, bl + (size_t)i * HC, xl);
    k_wmma_gemm<<<gGemm, T, 0, stream>>>(hbf, BpkR, zbias, xr);
    run_edge_phase(Wecv + (size_t)i * EDD * HC, aatt + (size_t)i * HH * CC);
    k_node_final<<<gNodeC, T, 0, stream>>>(accb, cb + (size_t)i * CC,
                                           bng + (size_t)(i + 1) * CC,
                                           bnb + (size_t)(i + 1) * CC,
                                           bnm + (size_t)(i + 1) * CC,
                                           bnv + (size_t)(i + 1) * CC,
                                           hbuf, hbuf, hbf);
  }

  // ---- global mean pool + MLP head ----
  k_zero<<<(int)(((long)GG * CC + T - 1) / T), T, 0, stream>>>(pooled, (long)GG * CC);
  k_zero<<<(int)((GG + T - 1) / T), T, 0, stream>>>(cnt, GG);
  k_pool<<<gNodeC, T, 0, stream>>>(hbuf, batch, pooled, cnt);
  k_head<<<(int)(((long)GG * 32 + T - 1) / T), T, 0, stream>>>(pooled, cnt, Wo1, bo1,
                                                               Wo2, bo2, (float*)d_out);
}
